// Block_59923383713845
// MI455X (gfx1250) — matrix operand order repaired and checked by reading, not compiled, not run on hardware
//
#include <hip/hip_runtime.h>
#include <hip/hip_bf16.h>
#include <math.h>

// ---------------------------------------------------------------------------
// Transformer block (B=2, N=2048, D=768, H=12, HD=64, HID=3072) for gfx1250.
// All GEMMs run on v_wmma_f32_16x16x32_f16 (f16 in, f32 accumulate).
// GEMM: 8 waves/block as 4x2, wave tile 32x64 -> block tile 128x128.
// ---------------------------------------------------------------------------

#define Bb   2
#define Nn   2048
#define Dd   768
#define Hh   12
#define HDd  64
#define HIDd 3072
#define ROWS (Bb * Nn)          // 4096
#define SCALE 0.125f            // 64^-0.5
#define EPSv 1e-5f

typedef __attribute__((ext_vector_type(16))) _Float16 v16h;
typedef __attribute__((ext_vector_type(8)))  _Float16 v8h;
typedef __attribute__((ext_vector_type(8)))  float    v8f;

static __device__ inline v8f zero8f() {
  v8f z;
  #pragma unroll
  for (int i = 0; i < 8; ++i) z[i] = 0.0f;
  return z;
}

static __device__ inline v8f wmma16x16x32(v16h a, v16h b, v8f c) {
  return __builtin_amdgcn_wmma_f32_16x16x32_f16(false, a, false, b,
                                                (short)0, c, false, false);
}

// A-fragment (16x32 f16, M x K). Row = lane&15. Lane group g = lane>>4:
//   halves 0..7  = A[m, g*8    .. g*8+7]
//   halves 8..15 = A[m, 16+g*8 .. 16+g*8+7]
static __device__ inline v16h load_a_frag(const _Float16* A, int lda, int lane) {
  const int g = (lane >> 4) & 1, m = lane & 15;
  const _Float16* p = A + (size_t)m * lda;
  v8h lo = *(const v8h*)(p + g * 8);
  v8h hi = *(const v8h*)(p + 16 + g * 8);
  v16h r;
  #pragma unroll
  for (int i = 0; i < 8; ++i) { r[i] = lo[i]; r[i + 8] = hi[i]; }
  return r;
}

// B-fragment (32x16 f16, K x N) taken from W stored N-major (row n, K contig):
//   lane (g, n): halves 0..7 = W[n, g*8 .. g*8+7], halves 8..15 = W[n, 16+g*8 .. 16+g*8+7]
static __device__ inline v16h load_b_frag(const _Float16* W, int ldw, int lane) {
  const int g = (lane >> 4) & 1, n = lane & 15;
  const _Float16* p = W + (size_t)n * ldw;
  v8h lo = *(const v8h*)(p + g * 8);
  v8h hi = *(const v8h*)(p + 16 + g * 8);
  v16h r;
  #pragma unroll
  for (int i = 0; i < 8; ++i) { r[i] = lo[i]; r[i + 8] = hi[i]; }
  return r;
}

// ---------------------------------------------------------------------------
// Elementwise helpers
// ---------------------------------------------------------------------------
__global__ void f32_to_f16_kernel(const float* __restrict__ in,
                                  _Float16* __restrict__ out, int n) {
  int i = blockIdx.x * blockDim.x + threadIdx.x;
  int stride = gridDim.x * blockDim.x;
  for (; i < n; i += stride) out[i] = (_Float16)in[i];
}

__global__ void build_qkv_bias_kernel(const float* __restrict__ qb,
                                      const float* __restrict__ vb,
                                      float* __restrict__ bias) {
  int i = blockIdx.x * blockDim.x + threadIdx.x;
  if (i >= 3 * Dd) return;
  float v = 0.0f;
  if (i < Dd) v = qb[i];
  else if (i >= 2 * Dd) v = vb[i - 2 * Dd];
  bias[i] = v;
}

// LayerNorm: one block per row of 768, writes f16.
__global__ __launch_bounds__(256)
void layernorm_kernel(const float* __restrict__ x, const float* __restrict__ g,
                      const float* __restrict__ b, _Float16* __restrict__ out) {
  __shared__ float s_sum[256], s_sq[256];
  const int row = blockIdx.x, tid = threadIdx.x;
  const float* xr = x + (size_t)row * Dd;
  float sum = 0.f, sq = 0.f, v[3];
  #pragma unroll
  for (int i = 0; i < 3; ++i) {
    v[i] = xr[tid + i * 256];
    sum += v[i]; sq += v[i] * v[i];
  }
  s_sum[tid] = sum; s_sq[tid] = sq;
  __syncthreads();
  for (int off = 128; off > 0; off >>= 1) {
    if (tid < off) { s_sum[tid] += s_sum[tid + off]; s_sq[tid] += s_sq[tid + off]; }
    __syncthreads();
  }
  const float mu  = s_sum[0] * (1.0f / Dd);
  const float var = s_sq[0] * (1.0f / Dd) - mu * mu;
  const float rstd = rsqrtf(var + EPSv);
  _Float16* orow = out + (size_t)row * Dd;
  #pragma unroll
  for (int i = 0; i < 3; ++i) {
    int c = tid + i * 256;
    orow[c] = (_Float16)((v[i] - mu) * rstd * g[c] + b[c]);
  }
}

// Repack qkv (ROWS x 3D, f16) -> Q,K as (b,h,n,d) and V^T as (b,h,d,n)
__global__ void repack_qkv_kernel(const _Float16* __restrict__ qkv,
                                  _Float16* __restrict__ q16,
                                  _Float16* __restrict__ k16,
                                  _Float16* __restrict__ vt16) {
  int idx = blockIdx.x * blockDim.x + threadIdx.x;
  if (idx >= Bb * Hh * Nn * HDd) return;
  int t = idx;
  const int d = t & 63; t >>= 6;
  const int n = t & (Nn - 1); t >>= 11;
  const int h = t % Hh;
  const int bb = t / Hh;
  const size_t src = ((size_t)(bb * Nn + n)) * (3 * Dd) + h * HDd + d;
  const size_t bh  = (size_t)(bb * Hh + h);
  q16 [(bh * Nn + n) * HDd + d] = qkv[src];
  k16 [(bh * Nn + n) * HDd + d] = qkv[src + Dd];
  vt16[(bh * HDd + d) * Nn + n] = qkv[src + 2 * Dd];
}

// ---------------------------------------------------------------------------
// WMMA GEMM:  C(MxN) = A(MxK,f16) * W(NxK,f16)^T + bias
// EPI 0: out16 = acc+bias          (f16)
// EPI 1: out16 = gelu(acc+bias)    (f16, exact erf gelu)
// EPI 2: out32 = resid + gamma[n]*(acc+bias)   (f32)
// 256 threads = 8 waves arranged 4x2; wave tile 32x64; block tile 128x128.
// ---------------------------------------------------------------------------
template <int EPI>
__global__ __launch_bounds__(256)
void gemm_wmma_kernel(const _Float16* __restrict__ A,
                      const _Float16* __restrict__ W,
                      const float* __restrict__ bias,
                      const float* __restrict__ resid,
                      const float* __restrict__ gamma,
                      _Float16* __restrict__ out16,
                      float* __restrict__ out32,
                      int Nc, int K) {
  const int lane = threadIdx.x & 31;
  const int wave = threadIdx.x >> 5;
  const int wrow = wave >> 1;       // 0..3
  const int wcol = wave & 1;        // 0..1
  const int row0 = blockIdx.x * 128 + wrow * 32;
  const int col0 = blockIdx.y * 128 + wcol * 64;

  v8f acc[2][4];
  #pragma unroll
  for (int i = 0; i < 2; ++i)
    #pragma unroll
    for (int j = 0; j < 4; ++j) acc[i][j] = zero8f();

  for (int kt = 0; kt < K; kt += 32) {
    v16h af[2];
    #pragma unroll
    for (int i = 0; i < 2; ++i)
      af[i] = load_a_frag(A + (size_t)(row0 + i * 16) * K + kt, K, lane);
    #pragma unroll
    for (int j = 0; j < 4; ++j) {
      v16h bf = load_b_frag(W + (size_t)(col0 + j * 16) * K + kt, K, lane);
      #pragma unroll
      for (int i = 0; i < 2; ++i)
        acc[i][j] = wmma16x16x32(af[i], bf, acc[i][j]);
    }
  }

  const int g = (lane >> 4) & 1, n = lane & 15;
  #pragma unroll
  for (int i = 0; i < 2; ++i) {
    #pragma unroll
    for (int j = 0; j < 4; ++j) {
      const int col = col0 + j * 16 + n;
      const float bv = bias[col];
      #pragma unroll
      for (int r = 0; r < 8; ++r) {
        const int row = row0 + i * 16 + g * 8 + r;
        float v = acc[i][j][r] + bv;
        const size_t o = (size_t)row * Nc + col;
        if (EPI == 0) {
          out16[o] = (_Float16)v;
        } else if (EPI == 1) {
          v = 0.5f * v * (1.0f + erff(v * 0.70710678118654752f));
          out16[o] = (_Float16)v;
        } else {
          out32[o] = resid[o] + gamma[col] * v;
        }
      }
    }
  }
}

// ---------------------------------------------------------------------------
// Streaming (flash-style) attention.
// grid = (N/64 qtiles, B*H); block = 128 (4 waves); each wave: 16 queries.
// ---------------------------------------------------------------------------
__global__ __launch_bounds__(128)
void attention_kernel(const _Float16* __restrict__ q16,
                      const _Float16* __restrict__ k16,
                      const _Float16* __restrict__ vt16,
                      const int* __restrict__ att_mask,
                      _Float16* __restrict__ o16) {
  __shared__ alignas(16) _Float16 lds_p[4][16][32];

  const int lane = threadIdx.x & 31;
  const int wave = threadIdx.x >> 5;
  const int bh   = blockIdx.y;             // b*H + h
  const int bb   = bh / Hh;
  const int h    = bh - bb * Hh;
  const int q0   = blockIdx.x * 64 + wave * 16;

  const _Float16* Qb  = q16  + ((size_t)bh * Nn) * HDd;
  const _Float16* Kb  = k16  + ((size_t)bh * Nn) * HDd;
  const _Float16* Vtb = vt16 + ((size_t)bh * HDd) * Nn;
  const int* maskb = att_mask + (size_t)bb * Nn;

  // Q fragments (16 queries x 64 dims = 2 K-chunks), reused for all key blocks
  v16h qf[2];
  #pragma unroll
  for (int c = 0; c < 2; ++c)
    qf[c] = load_a_frag(Qb + (size_t)q0 * HDd + c * 32, HDd, lane);

  const int g = (lane >> 4) & 1, ncol = lane & 15;

  float mrow[8], lrow[8];
  v8f oacc[4];
  #pragma unroll
  for (int r = 0; r < 8; ++r) { mrow[r] = -1e30f; lrow[r] = 0.0f; }
  #pragma unroll
  for (int t = 0; t < 4; ++t) oacc[t] = zero8f();

  for (int key0 = 0; key0 < Nn; key0 += 32) {
    // ---- S = Q * K^T (16 x 32 logits as two 16x16 tiles) ----
    v8f s0 = zero8f(), s1 = zero8f();
    #pragma unroll
    for (int c = 0; c < 2; ++c) {
      v16h bk0 = load_b_frag(Kb + (size_t)key0 * HDd + c * 32, HDd, lane);
      s0 = wmma16x16x32(qf[c], bk0, s0);
      v16h bk1 = load_b_frag(Kb + (size_t)(key0 + 16) * HDd + c * 32, HDd, lane);
      s1 = wmma16x16x32(qf[c], bk1, s1);
    }

    const int m0 = maskb[key0 + ncol];
    const int m1 = maskb[key0 + 16 + ncol];
    float sv0[8], sv1[8], rm[8];
    #pragma unroll
    for (int r = 0; r < 8; ++r) {
      sv0[r] = m0 ? s0[r] * SCALE : -1e30f;
      sv1[r] = m1 ? s1[r] * SCALE : -1e30f;
      rm[r]  = fmaxf(sv0[r], sv1[r]);
    }
    // row max across the 16 lanes of this half-wave group
    #pragma unroll
    for (int off = 1; off < 16; off <<= 1) {
      #pragma unroll
      for (int r = 0; r < 8; ++r)
        rm[r] = fmaxf(rm[r], __shfl_xor(rm[r], off, 32));
    }

    float p0[8], p1[8], rs[8];
    #pragma unroll
    for (int r = 0; r < 8; ++r) {
      const float mn    = fmaxf(mrow[r], rm[r]);
      const float alpha = __expf(mrow[r] - mn);
      mrow[r] = mn;
      lrow[r] *= alpha;
      #pragma unroll
      for (int t = 0; t < 4; ++t) oacc[t][r] *= alpha;
      p0[r] = __expf(sv0[r] - mn);
      p1[r] = __expf(sv1[r] - mn);
      rs[r] = p0[r] + p1[r];
    }
    #pragma unroll
    for (int off = 1; off < 16; off <<= 1) {
      #pragma unroll
      for (int r = 0; r < 8; ++r)
        rs[r] += __shfl_xor(rs[r], off, 32);
    }
    #pragma unroll
    for (int r = 0; r < 8; ++r) lrow[r] += rs[r];

    // ---- relayout P (C/D layout -> A fragment) through LDS, as f16 ----
    #pragma unroll
    for (int r = 0; r < 8; ++r) {
      lds_p[wave][g * 8 + r][ncol]      = (_Float16)p0[r];
      lds_p[wave][g * 8 + r][16 + ncol] = (_Float16)p1[r];
    }
    __builtin_amdgcn_wave_barrier();
    v16h pa = load_a_frag(&lds_p[wave][0][0], 32, lane);
    __builtin_amdgcn_wave_barrier();

    // ---- O += P * V  (V^T stored d-major, keys contiguous) ----
    #pragma unroll
    for (int t = 0; t < 4; ++t) {
      v16h bv = load_b_frag(Vtb + (size_t)(t * 16) * Nn + key0, Nn, lane);
      oacc[t] = wmma16x16x32(pa, bv, oacc[t]);
    }
  }

  // normalize and store to (b, n, h*64+d) f16
  #pragma unroll
  for (int r = 0; r < 8; ++r) {
    const float inv = 1.0f / lrow[r];
    const int   q   = q0 + g * 8 + r;
    _Float16* orow  = o16 + ((size_t)(bb * Nn + q)) * Dd + h * HDd;
    #pragma unroll
    for (int t = 0; t < 4; ++t)
      orow[t * 16 + ncol] = (_Float16)(oacc[t][r] * inv);
  }
}

// ---------------------------------------------------------------------------
// Host-side orchestration
// ---------------------------------------------------------------------------
extern "C" void kernel_launch(void* const* d_in, const int* in_sizes, int n_in,
                              void* d_out, int out_size, void* d_ws, size_t ws_size,
                              hipStream_t stream) {
  const float* x       = (const float*)d_in[0];
  const int*   mask    = (const int*)  d_in[1];
  const float* ln1_g   = (const float*)d_in[2];
  const float* ln1_b   = (const float*)d_in[3];
  const float* qkv_w   = (const float*)d_in[4];
  const float* q_bias  = (const float*)d_in[5];
  const float* v_bias  = (const float*)d_in[6];
  const float* proj_w  = (const float*)d_in[7];
  const float* proj_b  = (const float*)d_in[8];
  const float* ln2_g   = (const float*)d_in[9];
  const float* ln2_b   = (const float*)d_in[10];
  const float* fc1_w   = (const float*)d_in[11];
  const float* fc1_b   = (const float*)d_in[12];
  const float* fc2_w   = (const float*)d_in[13];
  const float* fc2_b   = (const float*)d_in[14];
  const float* gamma_1 = (const float*)d_in[15];
  const float* gamma_2 = (const float*)d_in[16];
  float* out = (float*)d_out;

  // workspace carve-up (~108 MB total)
  size_t off = 0;
  auto carve = [&](size_t bytes) -> void* {
    off = (off + 255) & ~(size_t)255;
    void* p = (char*)d_ws + off;
    off += bytes;
    return p;
  };
  _Float16* h16     = (_Float16*)carve((size_t)ROWS * Dd * 2);
  _Float16* h2_16   = (_Float16*)carve((size_t)ROWS * Dd * 2);
  _Float16* qkvw16  = (_Float16*)carve((size_t)3 * Dd * Dd * 2);
  _Float16* projw16 = (_Float16*)carve((size_t)Dd * Dd * 2);
  _Float16* fc1w16  = (_Float16*)carve((size_t)HIDd * Dd * 2);
  _Float16* fc2w16  = (_Float16*)carve((size_t)Dd * HIDd * 2);
  _Float16* qkv16   = (_Float16*)carve((size_t)ROWS * 3 * Dd * 2);
  _Float16* q16     = (_Float16*)carve((size_t)Bb * Hh * Nn * HDd * 2);
  _Float16* k16     = (_Float16*)carve((size_t)Bb * Hh * Nn * HDd * 2);
  _Float16* vt16    = (_Float16*)carve((size_t)Bb * Hh * Nn * HDd * 2);
  _Float16* o16     = (_Float16*)carve((size_t)ROWS * Dd * 2);
  _Float16* fc1o16  = (_Float16*)carve((size_t)ROWS * HIDd * 2);
  float*    x1      = (float*)   carve((size_t)ROWS * Dd * 4);
  float*    qkvbias = (float*)   carve((size_t)3 * Dd * 4);

  // 1) weight conversions (f32 -> f16)
  f32_to_f16_kernel<<<1024, 256, 0, stream>>>(qkv_w, qkvw16, 3 * Dd * Dd);
  f32_to_f16_kernel<<<1024, 256, 0, stream>>>(proj_w, projw16, Dd * Dd);
  f32_to_f16_kernel<<<1024, 256, 0, stream>>>(fc1_w, fc1w16, HIDd * Dd);
  f32_to_f16_kernel<<<1024, 256, 0, stream>>>(fc2_w, fc2w16, Dd * HIDd);
  build_qkv_bias_kernel<<<(3 * Dd + 255) / 256, 256, 0, stream>>>(q_bias, v_bias, qkvbias);

  // 2) LN1
  layernorm_kernel<<<ROWS, 256, 0, stream>>>(x, ln1_g, ln1_b, h16);

  // 3) QKV GEMM: (4096x768) x (2304x768)^T -> qkv16
  {
    dim3 grid(ROWS / 128, (3 * Dd) / 128);
    gemm_wmma_kernel<0><<<grid, 256, 0, stream>>>(h16, qkvw16, qkvbias,
                                                  nullptr, nullptr,
                                                  qkv16, nullptr, 3 * Dd, Dd);
  }

  // 4) repack Q/K/V^T
  repack_qkv_kernel<<<(Bb * Hh * Nn * HDd + 255) / 256, 256, 0, stream>>>(
      qkv16, q16, k16, vt16);

  // 5) attention
  {
    dim3 grid(Nn / 64, Bb * Hh);
    attention_kernel<<<grid, 128, 0, stream>>>(q16, k16, vt16, mask, o16);
  }

  // 6) proj GEMM + residual + gamma_1 -> x1 (f32)
  {
    dim3 grid(ROWS / 128, Dd / 128);
    gemm_wmma_kernel<2><<<grid, 256, 0, stream>>>(o16, projw16, proj_b,
                                                  x, gamma_1,
                                                  nullptr, x1, Dd, Dd);
  }

  // 7) LN2
  layernorm_kernel<<<ROWS, 256, 0, stream>>>(x1, ln2_g, ln2_b, h2_16);

  // 8) FC1 GEMM + exact GELU -> fc1o16
  {
    dim3 grid(ROWS / 128, HIDd / 128);
    gemm_wmma_kernel<1><<<grid, 256, 0, stream>>>(h2_16, fc1w16, fc1_b,
                                                  nullptr, nullptr,
                                                  fc1o16, nullptr, HIDd, Dd);
  }

  // 9) FC2 GEMM + residual + gamma_2 -> d_out (f32)
  {
    dim3 grid(ROWS / 128, Dd / 128);
    gemm_wmma_kernel<2><<<grid, 256, 0, stream>>>(fc1o16, fc2w16, fc2_b,
                                                  x1, gamma_2,
                                                  nullptr, out, Dd, HIDd);
  }
}
